// Model_58196806861249
// MI455X (gfx1250) — matrix operand order repaired and checked by reading, not compiled, not run on hardware
//
#include <hip/hip_runtime.h>
#include <hip/hip_bf16.h>

#define B_      64
#define T_      256
#define CELL_   256
#define GATES_  1024
#define LABELS_ 128
#define NLAYERS 8
#define NWG     8

#define KP   520   // LDS row stride (f16) for LSTM kernel (512 + pad)
#define ZP   132   // LDS z row stride (f32) (128 + pad)
#define KP2  264   // LDS row stride (f16) for final kernel (256 + pad)

typedef __attribute__((ext_vector_type(16))) _Float16 v16h;
typedef __attribute__((ext_vector_type(8)))  _Float16 v8h;
typedef __attribute__((ext_vector_type(8)))  float    v8f;

__device__ __forceinline__ float sigmoidf_(float x) {
    return 1.0f / (1.0f + __expf(-x));
}

// One 16x16x32 f16 WMMA step. ap points at this lane's A row (half-wave K base
// already applied: K {kb..kb+7} contiguous, K {kb+16..kb+23} at +16). bp points
// at this lane's B column (same layout: K {kb..kb+7}, K {kb+16..kb+23} at +16).
__device__ __forceinline__ v8f wmma_step(const _Float16* ap, const _Float16* bp, v8f c) {
    v8h a0 = *(const v8h*)(ap);
    v8h a1 = *(const v8h*)(ap + 16);
    v8h b0 = *(const v8h*)(bp);
    v8h b1 = *(const v8h*)(bp + 16);
    v16h a, b;
#pragma unroll
    for (int e = 0; e < 8; ++e) { a[e] = a0[e]; a[e + 8] = a1[e]; b[e] = b0[e]; b[e + 8] = b1[e]; }
    return __builtin_amdgcn_wmma_f32_16x16x32_f16(false, a, false, b, (short)0, c, false, false);
}

// ---------------------------------------------------------------------------
// Persistent fused-LSTM layer kernel. 8 workgroups x 256 threads (8 waves).
// WG w owns hidden columns [w*32, w*32+32) and the 4 matching gate column
// blocks. The fused weight slice [Wi;Wh] (512 x 128 f16, stored transposed
// [n][k]) stays in LDS for all 256 timesteps.
// ---------------------------------------------------------------------------
__global__ __launch_bounds__(256) void lstm_layer_kernel(
    const _Float16* __restrict__ x,        // [B][T][256] f16 layer input
    const _Float16* __restrict__ Wi,       // [256][1024] f16
    const _Float16* __restrict__ Wh,       // [256][1024] f16
    const float*    __restrict__ bias,     // [1024] f32
    const int*      __restrict__ word_ids, // [B][T] mask source
    _Float16*       __restrict__ hbuf,     // [2][B][256] ping-pong h state
    _Float16*       __restrict__ yout,     // [B][T][256] layer output
    unsigned*       __restrict__ bar,      // grid barrier counter (zeroed)
    int reverse)
{
    extern __shared__ char smem[];
    _Float16* Ws = (_Float16*)smem;          // [128][KP]  weights, n-major
    _Float16* hx = Ws + 128 * KP;            // [64][KP]   [x_t ; h] per batch row
    float*    zb = (float*)(hx + 64 * KP);   // [64][ZP]   gate pre-activations

    const int w    = blockIdx.x;
    const int tid  = threadIdx.x;
    const int lane = tid & 31;
    const int wv   = tid >> 5;

    // Preload fused [Wi;Wh] slice transposed to [n][k] (once per layer).
    for (int idx = tid; idx < 128 * 512; idx += 256) {
        int n  = idx & 127;
        int k  = idx >> 7;
        int g  = n >> 5;
        int gc = g * CELL_ + w * 32 + (n & 31);
        _Float16 v = (k < 256) ? Wi[(size_t)k * GATES_ + gc]
                               : Wh[(size_t)(k - 256) * GATES_ + gc];
        Ws[n * KP + k] = v;
    }

    const int j = tid & 31;                      // hidden col (local) this thread owns
    const float bI = bias[0 * CELL_ + w * 32 + j];
    const float bF = bias[1 * CELL_ + w * 32 + j];
    const float bC = bias[2 * CELL_ + w * 32 + j];
    const float bO = bias[3 * CELL_ + w * 32 + j];

    float creg[8];                               // c state: thread owns (b, j) pairs
#pragma unroll
    for (int q = 0; q < 8; ++q) creg[q] = 0.0f;

    __syncthreads();

    for (int t = 0; t < T_; ++t) {
        const int tsrc = reverse ? (T_ - 1 - t) : t;
        const _Float16* hcur = hbuf + (size_t)(t & 1) * (B_ * CELL_);
        _Float16*       hnxt = hbuf + (size_t)((t & 1) ^ 1) * (B_ * CELL_);

        // Stage x_t and h into LDS: hx[b][0:256]=x_t, hx[b][256:512]=h.
        for (int idx = tid; idx < (B_ * CELL_) / 8; idx += 256) {
            int b  = idx >> 5;
            int kc = (idx & 31) * 8;
            *(v8h*)&hx[b * KP + kc]       = *(const v8h*)&x[((size_t)(b * T_ + tsrc)) * CELL_ + kc];
            *(v8h*)&hx[b * KP + 256 + kc] = *(const v8h*)&hcur[b * CELL_ + kc];
        }
        // Prefetch next timestep's activations into cache.
        if (t + 1 < T_) {
            int tn2 = reverse ? (T_ - 2 - t) : (t + 1);
            __builtin_prefetch(&x[((size_t)((tid >> 2) * T_ + tn2)) * CELL_ + (tid & 3) * 64], 0, 1);
        }
        __syncthreads();

        // z = [x_t ; h] @ [Wi ; Wh] for this WG's 128 gate columns.
        // 4 M-tiles x 8 N-tiles = 32 tiles; 8 waves x 4 tiles; K=512 -> 16 WMMAs each.
#pragma unroll
        for (int tt = 0; tt < 4; ++tt) {
            const int tile = (wv << 2) + tt;
            const int tm   = (tile & 3) << 4;
            const int tn   = (tile >> 2) << 4;
            const int arow = tm + (lane & 15);
            const int kbA  = (lane & 16) ? 8 : 0;
            const int bcol = tn + (lane & 15);
            const int kbB  = (lane & 16) ? 8 : 0;
            const _Float16* ap = &hx[arow * KP + kbA];
            const _Float16* bp = &Ws[bcol * KP + kbB];
            v8f c = {};
#pragma unroll
            for (int kk = 0; kk < 512; kk += 32)
                c = wmma_step(ap + kk, bp + kk, c);
            const int zr = tm + ((lane & 16) ? 8 : 0);
            const int zc = tn + (lane & 15);
#pragma unroll
            for (int r = 0; r < 8; ++r) zb[(zr + r) * ZP + zc] = c[r];
        }
        __syncthreads();

        // Gate math + masked state update (Keras semantics: carry h,c when masked,
        // emit the carried h).
#pragma unroll
        for (int q = 0; q < 8; ++q) {
            const int b = (tid >> 5) + 8 * q;
            float zi = zb[b * ZP +  0 + j] + bI;
            float zf = zb[b * ZP + 32 + j] + bF;
            float zg = zb[b * ZP + 64 + j] + bC;
            float zo = zb[b * ZP + 96 + j] + bO;
            float ig = sigmoidf_(zi);
            float fg = sigmoidf_(zf);
            float og = sigmoidf_(zo);
            float cn = fg * creg[q] + ig * tanhf(zg);
            float hn = og * tanhf(cn);
            bool  m  = word_ids[b * T_ + tsrc] != 0;
            float hp = (float)hx[b * KP + 256 + w * 32 + j];
            float ho = m ? hn : hp;
            creg[q]  = m ? cn : creg[q];
            hnxt[b * CELL_ + w * 32 + j]                      = (_Float16)ho;
            yout[((size_t)(b * T_ + t)) * CELL_ + w * 32 + j] = (_Float16)ho;
        }

        // Device-wide barrier: all 8 WGs (1 per WGP, forced by LDS footprint)
        // must see the new h before the next step. Monotonic counter.
        __syncthreads();
        __threadfence();
        if (tid == 0) {
            atomicAdd(bar, 1u);
            const unsigned target = (unsigned)NWG * (unsigned)(t + 1);
            while (*(volatile unsigned*)bar < target) __builtin_amdgcn_s_sleep(1);
        }
        __syncthreads();
    }
}

// ---------------------------------------------------------------------------
// Final dense + softmax: [16384 x 256] x [256 x 128] + bias, softmax over 128.
// One block computes 64 rows. A tile and Wd^T staged in LDS; WMMA tiled.
// ---------------------------------------------------------------------------
__global__ __launch_bounds__(256) void dense_softmax_kernel(
    const _Float16* __restrict__ A,     // [BT][256] f16 (= outs6+outs7)
    const _Float16* __restrict__ Wd16,  // [256][128] f16
    const float*    __restrict__ bd,    // [128]
    float*          __restrict__ out)   // [BT][128]
{
    extern __shared__ char smem[];
    _Float16* Al = (_Float16*)smem;            // [64][KP2]
    _Float16* Wl = Al + 64 * KP2;              // [128][KP2] label-major
    float*    zb = (float*)(Wl + 128 * KP2);   // [64][ZP]
    const int tid = threadIdx.x, lane = tid & 31, wv = tid >> 5;
    const size_t row0 = (size_t)blockIdx.x * 64;

    for (int idx = tid; idx < (64 * 256) / 8; idx += 256) {
        int r  = idx >> 5;
        int kc = (idx & 31) * 8;
        *(v8h*)&Al[r * KP2 + kc] = *(const v8h*)&A[(row0 + r) * CELL_ + kc];
    }
    for (int idx = tid; idx < 128 * 256; idx += 256) {
        int n = idx & 127;
        int k = idx >> 7;
        Wl[n * KP2 + k] = Wd16[(size_t)k * LABELS_ + n];
    }
    __syncthreads();

#pragma unroll
    for (int tt = 0; tt < 4; ++tt) {
        const int tile = (wv << 2) + tt;
        const int tm   = (tile & 3) << 4;
        const int tn   = (tile >> 2) << 4;
        const _Float16* ap = &Al[(tm + (lane & 15)) * KP2 + ((lane & 16) ? 8 : 0)];
        const _Float16* bp = &Wl[(tn + (lane & 15)) * KP2 + ((lane & 16) ? 8 : 0)];
        v8f c = {};
#pragma unroll
        for (int kk = 0; kk < 256; kk += 32)
            c = wmma_step(ap + kk, bp + kk, c);
        const int zr = tm + ((lane & 16) ? 8 : 0);
        const int zc = tn + (lane & 15);
#pragma unroll
        for (int r = 0; r < 8; ++r) zb[(zr + r) * ZP + zc] = c[r];
    }
    __syncthreads();

    if (tid < 64) {
        const int r = tid;
        float mx = -3.4e38f;
        for (int n = 0; n < LABELS_; ++n) {
            float v = zb[r * ZP + n] + bd[n];
            zb[r * ZP + n] = v;
            mx = fmaxf(mx, v);
        }
        float s = 0.0f;
        for (int n = 0; n < LABELS_; ++n) {
            float e = __expf(zb[r * ZP + n] - mx);
            zb[r * ZP + n] = e;
            s += e;
        }
        float inv = 1.0f / s;
        for (int n = 0; n < LABELS_; ++n)
            out[(row0 + r) * LABELS_ + n] = zb[r * ZP + n] * inv;
    }
}

// ---------------------------------------------------------------------------
// Small helper kernels.
// ---------------------------------------------------------------------------
__global__ void cvt_f32_f16_kernel(const float* __restrict__ in, _Float16* __restrict__ o, int n) {
    for (int i = blockIdx.x * blockDim.x + threadIdx.x; i < n; i += gridDim.x * blockDim.x)
        o[i] = (_Float16)in[i];
}

__global__ void embed_concat_kernel(const int* __restrict__ wid, const int* __restrict__ pid,
                                    const float* __restrict__ e1, const float* __restrict__ e2,
                                    _Float16* __restrict__ xo) {
    const int total = B_ * T_ * CELL_;
    for (int i = blockIdx.x * blockDim.x + threadIdx.x; i < total; i += gridDim.x * blockDim.x) {
        int bt = i >> 8;
        int k  = i & 255;
        float v = (k < 128) ? e1[(size_t)wid[bt] * 128 + k]
                            : e2[(size_t)pid[bt] * 128 + (k - 128)];
        xo[i] = (_Float16)v;
    }
}

__global__ void add16_kernel(const _Float16* __restrict__ a, const _Float16* __restrict__ b,
                             _Float16* __restrict__ o, int n) {
    for (int i = blockIdx.x * blockDim.x + threadIdx.x; i < n; i += gridDim.x * blockDim.x)
        o[i] = (_Float16)((float)a[i] + (float)b[i]);
}

// ---------------------------------------------------------------------------
extern "C" void kernel_launch(void* const* d_in, const int* in_sizes, int n_in,
                              void* d_out, int out_size, void* d_ws, size_t ws_size,
                              hipStream_t stream)
{
    (void)in_sizes; (void)n_in; (void)out_size; (void)ws_size;
    const int*   word_ids = (const int*)  d_in[0];
    const int*   pred_ids = (const int*)  d_in[1];
    const float* emb1     = (const float*)d_in[2];
    const float* emb2     = (const float*)d_in[3];
    const float* Wis      = (const float*)d_in[4];
    const float* Whs      = (const float*)d_in[5];
    const float* bs       = (const float*)d_in[6];
    const float* Wd       = (const float*)d_in[7];
    const float* bd       = (const float*)d_in[8];

    char*  ws  = (char*)d_ws;
    size_t off = 0;
    auto take = [&](size_t bytes) -> void* {
        void* p = (void*)(ws + off);
        off = (off + bytes + 255) & ~(size_t)255;
        return p;
    };

    const size_t BT = (size_t)B_ * T_;
    _Float16* x16 = (_Float16*)take(BT * CELL_ * sizeof(_Float16));
    _Float16* ob[3];
    for (int i = 0; i < 3; ++i) ob[i] = (_Float16*)take(BT * CELL_ * sizeof(_Float16));
    _Float16* Wi16 = (_Float16*)take((size_t)NLAYERS * CELL_ * GATES_ * sizeof(_Float16));
    _Float16* Wh16 = (_Float16*)take((size_t)NLAYERS * CELL_ * GATES_ * sizeof(_Float16));
    _Float16* Wd16 = (_Float16*)take((size_t)CELL_ * LABELS_ * sizeof(_Float16));
    _Float16* hbuf = (_Float16*)take((size_t)2 * B_ * CELL_ * sizeof(_Float16));
    unsigned* bar  = (unsigned*)take(256);

    cvt_f32_f16_kernel<<<2048, 256, 0, stream>>>(Wis, Wi16, NLAYERS * CELL_ * GATES_);
    cvt_f32_f16_kernel<<<2048, 256, 0, stream>>>(Whs, Wh16, NLAYERS * CELL_ * GATES_);
    cvt_f32_f16_kernel<<<64,   256, 0, stream>>>(Wd,  Wd16, CELL_ * LABELS_);
    embed_concat_kernel<<<2048, 256, 0, stream>>>(word_ids, pred_ids, emb1, emb2, x16);

    const size_t lstmSmem = (size_t)128 * KP * sizeof(_Float16)
                          + (size_t)64  * KP * sizeof(_Float16)
                          + (size_t)64  * ZP * sizeof(float);      // ~228 KB / WGP
    for (int i = 0; i < NLAYERS; ++i) {
        if (i == 1) {
            hipMemcpyAsync(x16, ob[0], BT * CELL_ * sizeof(_Float16),
                           hipMemcpyDeviceToDevice, stream);
        } else if (i >= 2) {
            add16_kernel<<<2048, 256, 0, stream>>>(ob[(i - 1) % 3], ob[(i - 2) % 3],
                                                   x16, (int)(BT * CELL_));
        }
        hipMemsetAsync(hbuf, 0, (size_t)2 * B_ * CELL_ * sizeof(_Float16), stream);
        hipMemsetAsync(bar, 0, 256, stream);
        lstm_layer_kernel<<<NWG, 256, lstmSmem, stream>>>(
            x16,
            Wi16 + (size_t)i * CELL_ * GATES_,
            Wh16 + (size_t)i * CELL_ * GATES_,
            bs + (size_t)i * GATES_,
            word_ids, hbuf, ob[i % 3], bar, i & 1);
    }

    // outs[6] -> ob[0], outs[7] -> ob[1]
    add16_kernel<<<2048, 256, 0, stream>>>(ob[0], ob[1], x16, (int)(BT * CELL_));
    const size_t finSmem = (size_t)64  * KP2 * sizeof(_Float16)
                         + (size_t)128 * KP2 * sizeof(_Float16)
                         + (size_t)64  * ZP  * sizeof(float);      // ~132 KB
    dense_softmax_kernel<<<(int)(BT / 64), 256, finSmem, stream>>>(x16, Wd16, bd, (float*)d_out);
}
